// RecurrentGCN_37769942401406
// MI455X (gfx1250) — matrix operand order repaired and checked by reading, not compiled, not run on hardware
//
#include <hip/hip_runtime.h>
#include <hip/hip_bf16.h>

typedef __attribute__((ext_vector_type(16))) _Float16 v16h;
typedef __attribute__((ext_vector_type(8)))  float    v8f;

#define TP 12   // time periods
#define CH 32   // hidden channels

// ---------------- graph aggregation ----------------

__global__ __launch_bounds__(256) void k_init_deg(float* __restrict__ deg, int n) {
  int i = blockIdx.x * blockDim.x + threadIdx.x;
  if (i < n) deg[i] = 1.0f;  // self-loop weight (fill value 1)
}

__global__ __launch_bounds__(256) void k_deg_scatter(const int* __restrict__ ei,
                                                     const float* __restrict__ ew,
                                                     float* __restrict__ deg, int ecount) {
  int i = blockIdx.x * blockDim.x + threadIdx.x;
  if (i >= ecount) return;
  int d = ei[ecount + i];
  __hip_atomic_fetch_add(&deg[d], ew[i], __ATOMIC_RELAXED, __HIP_MEMORY_SCOPE_AGENT);
}

__global__ __launch_bounds__(256) void k_dinv_selfloop(const float* __restrict__ deg,
                                                       const float* __restrict__ x,
                                                       float* __restrict__ dinv,
                                                       float* __restrict__ agg, int n) {
  int i = blockIdx.x * blockDim.x + threadIdx.x;
  if (i >= n) return;
  float dg = deg[i];
  float dv = dg > 0.f ? rsqrtf(dg) : 0.f;
  dinv[i] = dv;
  float s = dv * dv;  // self-loop: dinv[n]*1*dinv[n]
  const float4* xr = (const float4*)(x + (size_t)i * TP);
  float4* ar = (float4*)(agg + (size_t)i * TP);
#pragma unroll
  for (int q = 0; q < 3; ++q) {
    float4 v = xr[q];
    v.x *= s; v.y *= s; v.z *= s; v.w *= s;
    ar[q] = v;
  }
}

__global__ __launch_bounds__(256) void k_edge_scatter(const int* __restrict__ ei,
                                                      const float* __restrict__ ew,
                                                      const float* __restrict__ dinv,
                                                      const float* __restrict__ x,
                                                      float* __restrict__ agg, int ecount) {
  int i = blockIdx.x * blockDim.x + threadIdx.x;
  if (i >= ecount) return;
  int s = ei[i];
  int d = ei[ecount + i];
  float nrm = dinv[s] * ew[i] * dinv[d];
  const float4* xr = (const float4*)(x + (size_t)s * TP);  // 48B rows, 16B aligned
  float4 x0 = xr[0], x1 = xr[1], x2 = xr[2];
  float* ad = agg + (size_t)d * TP;
#define ATADD(off, v) __hip_atomic_fetch_add(ad + (off), nrm * (v), __ATOMIC_RELAXED, __HIP_MEMORY_SCOPE_AGENT)
  ATADD(0, x0.x);  ATADD(1, x0.y);  ATADD(2, x0.z);  ATADD(3, x0.w);
  ATADD(4, x1.x);  ATADD(5, x1.y);  ATADD(6, x1.z);  ATADD(7, x1.w);
  ATADD(8, x2.x);  ATADD(9, x2.y);  ATADD(10, x2.z); ATADD(11, x2.w);
#undef ATADD
}

// ---------------- WMMA finale ----------------
// One wave handles 16 nodes. For each t: build A = cz/ch (16x32 f16) from scalar
// agg, do D = A x lw^T (+bias in C) via v_wmma_f32_16x16x32_f16 (two 16-col
// o-tiles per gate), apply sigmoid/tanh, accumulate probs[t]-weighted Hp.
// Then ReLU + head dot-product with a 16-lane shfl_xor reduction.

__device__ __forceinline__ float sigf(float xv) { return 1.f / (1.f + __expf(-xv)); }
__device__ __forceinline__ float tanhfast(float xv) { return 2.f / (1.f + __expf(-2.f * xv)) - 1.f; }

__global__ __launch_bounds__(256) void k_finale_wmma(
    const float* __restrict__ agg,
    const float* __restrict__ w_z, const float* __restrict__ b_z,
    const float* __restrict__ w_h, const float* __restrict__ b_h,
    const float* __restrict__ lw_z, const float* __restrict__ lb_z,
    const float* __restrict__ lw_h, const float* __restrict__ lb_h,
    const float* __restrict__ att, const float* __restrict__ head_w,
    const float* __restrict__ head_b, float* __restrict__ out, int n_tiles) {
  const int wave = (blockIdx.x * blockDim.x + threadIdx.x) >> 5;
  const int lane = threadIdx.x & 31;
  if (wave >= n_tiles) return;  // wave-uniform; EXEC stays all-ones past here
  const int base = wave * 16;
  const int lm = lane & 15;   // column / row-in-half index
  const int hi = lane >> 4;   // which half of the wave

  // softmax(att), redundantly per lane (T=12)
  float pr[TP];
  float mx = -1e30f;
#pragma unroll
  for (int t = 0; t < TP; ++t) { pr[t] = att[t]; mx = fmaxf(mx, pr[t]); }
  float se = 0.f;
#pragma unroll
  for (int t = 0; t < TP; ++t) { pr[t] = __expf(pr[t] - mx); se += pr[t]; }
  const float sinv = 1.f / se;
#pragma unroll
  for (int t = 0; t < TP; ++t) pr[t] *= sinv;

  // B tiles: B[k][o] = lw[o][k], contiguous k order used here (A matches):
  // lanes 0-15 hold K=0..15 (halfs 0..15), lanes 16-31 hold K=16..31; o = lane%16 (+16 for tile 1)
  v16h Bz0, Bz1, Bh0, Bh1;
#pragma unroll
  for (int hh = 0; hh < 16; ++hh) {
    int k = hh + hi * 16;
    Bz0[hh] = (_Float16)lw_z[(size_t)lm * 2 * CH + k];
    Bz1[hh] = (_Float16)lw_z[(size_t)(lm + 16) * 2 * CH + k];
    Bh0[hh] = (_Float16)lw_h[(size_t)lm * 2 * CH + k];
    Bh1[hh] = (_Float16)lw_h[(size_t)(lm + 16) * 2 * CH + k];
  }

  // A-side per-lane K mapping (matches B): halfs 0..15 -> K=hh (+16 in hi half)
  float wzk[16], bzk[16], whk[16], bhk[16];
#pragma unroll
  for (int hh = 0; hh < 16; ++hh) {
    int k = hh + hi * 16;
    wzk[hh] = w_z[k]; bzk[hh] = b_z[k];
    whk[hh] = w_h[k]; bhk[hh] = b_h[k];
  }

  // bias-preloaded C accumulators (per lane the whole column shares one bias)
  const float lbz0 = lb_z[lm], lbz1 = lb_z[lm + 16];
  const float lbh0 = lb_h[lm], lbh1 = lb_h[lm + 16];
  v8f Cz0, Cz1, Ch0, Ch1;
#pragma unroll
  for (int j = 0; j < 8; ++j) { Cz0[j] = lbz0; Cz1[j] = lbz1; Ch0[j] = lbh0; Ch1[j] = lbh1; }

  // agg row for this lane's node (M = lane%16)
  const float4* ar = (const float4*)(agg + (size_t)(base + lm) * TP);
  float4 a0 = ar[0], a1 = ar[1], a2 = ar[2];
  float aggv[TP] = {a0.x, a0.y, a0.z, a0.w, a1.x, a1.y, a1.z, a1.w, a2.x, a2.y, a2.z, a2.w};

  float hacc0[8] = {0, 0, 0, 0, 0, 0, 0, 0};
  float hacc1[8] = {0, 0, 0, 0, 0, 0, 0, 0};

#pragma unroll
  for (int t = 0; t < TP; ++t) {
    const float av = aggv[t];
    v16h Az, Ah;
#pragma unroll
    for (int hh = 0; hh < 16; ++hh) {
      Az[hh] = (_Float16)fmaf(av, wzk[hh], bzk[hh]);
      Ah[hh] = (_Float16)fmaf(av, whk[hh], bhk[hh]);
    }
    v8f Dz0 = __builtin_amdgcn_wmma_f32_16x16x32_f16(false, Az, false, Bz0, (short)0, Cz0, false, false);
    v8f Dz1 = __builtin_amdgcn_wmma_f32_16x16x32_f16(false, Az, false, Bz1, (short)0, Cz1, false, false);
    v8f Dh0 = __builtin_amdgcn_wmma_f32_16x16x32_f16(false, Ah, false, Bh0, (short)0, Ch0, false, false);
    v8f Dh1 = __builtin_amdgcn_wmma_f32_16x16x32_f16(false, Ah, false, Bh1, (short)0, Ch1, false, false);
    const float p = pr[t];
#pragma unroll
    for (int j = 0; j < 8; ++j) {
      float hp0 = (1.f - sigf(Dz0[j])) * tanhfast(Dh0[j]);
      float hp1 = (1.f - sigf(Dz1[j])) * tanhfast(Dh1[j]);
      hacc0[j] = fmaf(p, hp0, hacc0[j]);
      hacc1[j] = fmaf(p, hp1, hacc1[j]);
    }
  }

  // head: out[n] = sum_o head_w[o]*relu(h[n,o]) + head_b
  const float hw0 = head_w[lm], hw1 = head_w[lm + 16];
  const float hb = head_b[0];
  float part[8];
#pragma unroll
  for (int j = 0; j < 8; ++j)
    part[j] = hw0 * fmaxf(hacc0[j], 0.f) + hw1 * fmaxf(hacc1[j], 0.f);
#pragma unroll
  for (int j = 0; j < 8; ++j) {
    part[j] += __shfl_xor(part[j], 1, 32);
    part[j] += __shfl_xor(part[j], 2, 32);
    part[j] += __shfl_xor(part[j], 4, 32);
    part[j] += __shfl_xor(part[j], 8, 32);
  }
  if (lm == 0) {
    int row0 = hi * 8;  // lanes 0-15 carry rows 0-7, lanes 16-31 rows 8-15
#pragma unroll
    for (int j = 0; j < 8; ++j) out[base + row0 + j] = part[j] + hb;
  }
}

// scalar tail for N % 16 != 0 (not hit for N=100000, kept for generality)
__global__ void k_tail(const float* __restrict__ agg,
                       const float* __restrict__ w_z, const float* __restrict__ b_z,
                       const float* __restrict__ w_h, const float* __restrict__ b_h,
                       const float* __restrict__ lw_z, const float* __restrict__ lb_z,
                       const float* __restrict__ lw_h, const float* __restrict__ lb_h,
                       const float* __restrict__ att, const float* __restrict__ head_w,
                       const float* __restrict__ head_b, float* __restrict__ out,
                       int start, int n) {
  int node = start + blockIdx.x * blockDim.x + threadIdx.x;
  if (node >= n) return;
  float pr[TP]; float mx = -1e30f;
  for (int t = 0; t < TP; ++t) { pr[t] = att[t]; mx = fmaxf(mx, pr[t]); }
  float se = 0.f;
  for (int t = 0; t < TP; ++t) { pr[t] = __expf(pr[t] - mx); se += pr[t]; }
  for (int t = 0; t < TP; ++t) pr[t] /= se;
  float h[CH];
  for (int o = 0; o < CH; ++o) h[o] = 0.f;
  for (int t = 0; t < TP; ++t) {
    float a = agg[(size_t)node * TP + t];
    for (int o = 0; o < CH; ++o) {
      float zac = lb_z[o], hac = lb_h[o];
      for (int k = 0; k < CH; ++k) {
        zac += (a * w_z[k] + b_z[k]) * lw_z[(size_t)o * 2 * CH + k];
        hac += (a * w_h[k] + b_h[k]) * lw_h[(size_t)o * 2 * CH + k];
      }
      h[o] += pr[t] * (1.f - sigf(zac)) * tanhfast(hac);
    }
  }
  float acc = head_b[0];
  for (int o = 0; o < CH; ++o) acc += head_w[o] * fmaxf(h[o], 0.f);
  out[node] = acc;
}

extern "C" void kernel_launch(void* const* d_in, const int* in_sizes, int n_in,
                              void* d_out, int out_size, void* d_ws, size_t ws_size,
                              hipStream_t stream) {
  const float* x      = (const float*)d_in[0];
  const int*   ei     = (const int*)d_in[1];
  const float* ew     = (const float*)d_in[2];
  const float* w_z    = (const float*)d_in[3];
  const float* b_z    = (const float*)d_in[4];
  const float* w_h    = (const float*)d_in[7];
  const float* b_h    = (const float*)d_in[8];
  const float* lw_z   = (const float*)d_in[9];
  const float* lb_z   = (const float*)d_in[10];
  const float* lw_h   = (const float*)d_in[13];
  const float* lb_h   = (const float*)d_in[14];
  const float* att    = (const float*)d_in[15];
  const float* head_w = (const float*)d_in[16];
  const float* head_b = (const float*)d_in[17];

  const int N = in_sizes[0] / TP;
  const int E = in_sizes[2];

  float* deg  = (float*)d_ws;       // N floats
  float* dinv = deg + N;            // N floats
  float* agg  = dinv + N;           // N*TP floats (offset 2N*4B, 16B aligned)

  k_init_deg<<<(N + 255) / 256, 256, 0, stream>>>(deg, N);
  k_deg_scatter<<<(E + 255) / 256, 256, 0, stream>>>(ei, ew, deg, E);
  k_dinv_selfloop<<<(N + 255) / 256, 256, 0, stream>>>(deg, x, dinv, agg, N);
  k_edge_scatter<<<(E + 255) / 256, 256, 0, stream>>>(ei, ew, dinv, x, agg, E);

  const int tiles = N / 16;
  if (tiles > 0) {
    const int threads = tiles * 32;  // one wave32 per 16-node tile
    k_finale_wmma<<<(threads + 255) / 256, 256, 0, stream>>>(
        agg, w_z, b_z, w_h, b_h, lw_z, lb_z, lw_h, lb_h,
        att, head_w, head_b, (float*)d_out, tiles);
  }
  const int rem = N - tiles * 16;
  if (rem > 0) {
    k_tail<<<1, 64, 0, stream>>>(agg, w_z, b_z, w_h, b_h, lw_z, lb_z, lw_h, lb_h,
                                 att, head_w, head_b, (float*)d_out, tiles * 16, N);
  }
}
